// capsnet_59382217835207
// MI455X (gfx1250) — matrix operand order repaired and checked by reading, not compiled, not run on hardware
//
#include <hip/hip_runtime.h>
#include <math.h>
#include <stdint.h>

typedef __attribute__((ext_vector_type(16))) _Float16 v16h;
typedef __attribute__((ext_vector_type(8)))  _Float16 v8h;
typedef __attribute__((ext_vector_type(8)))  float    v8f;
typedef __attribute__((ext_vector_type(4)))  unsigned v4u;
typedef __attribute__((ext_vector_type(8)))  int      v8i;
typedef __attribute__((ext_vector_type(4)))  int      v4i;

#if __has_builtin(__builtin_amdgcn_tensor_load_to_lds)
#define HAVE_TDM 1
#else
#define HAVE_TDM 0
#endif

#define STAGE_CHUNKS 9                    // 81*(Cin/32) is always divisible by 9
#define STAGE_BYTES  (STAGE_CHUNKS * 1024)  // 9 chunks x 1KB B-fragment

// ---------------------------------------------------------------------------
// Weight repack: OIHW f32 -> WMMA-B-fragment-swizzled f16.
//   K = kt*32 + (l>>4)*8 + (j&7) + (j>>3)*16   (lanes 0-15: K 0..7,16..23; lanes 16-31: K 8..15,24..31)
//   N = nt*16 + (l&15)
// K ordering is (kh, kw, cin) to match the implicit-GEMM A-patch walk.
// ---------------------------------------------------------------------------
__global__ void repack_conv_w_k(const float* __restrict__ w, _Float16* __restrict__ frag,
                                int Cin, int Cout)
{
    size_t idx = (size_t)blockIdx.x * blockDim.x + threadIdx.x;
    size_t total = (size_t)81 * Cin * Cout;
    if (idx >= total) return;
    int j = (int)(idx & 15);
    int l = (int)((idx >> 4) & 31);
    size_t t2 = idx >> 9;                      // kt*Ntiles + nt
    int Ntiles = Cout >> 4;
    int nt = (int)(t2 % Ntiles);
    int kt = (int)(t2 / Ntiles);
    int K = kt * 32 + (l >> 4) * 8 + (j & 7) + (j >> 3) * 16;
    int n = nt * 16 + (l & 15);
    int cin = K % Cin;
    int kk  = K / Cin;                          // kh*9 + kw
    int kw  = kk % 9, kh = kk / 9;
    frag[idx] = (_Float16)w[(((size_t)n * Cin + cin) * 9 + kh) * 9 + kw];
}

// Dense [N][K] f32 -> swizzled f16 B fragments
__global__ void repack_dense_w_k(const float* __restrict__ w, _Float16* __restrict__ frag,
                                 int K, int N)
{
    size_t idx = (size_t)blockIdx.x * blockDim.x + threadIdx.x;
    size_t total = (size_t)K * N;
    if (idx >= total) return;
    int j = (int)(idx & 15);
    int l = (int)((idx >> 4) & 31);
    size_t t2 = idx >> 9;
    int Ntiles = N >> 4;
    int nt = (int)(t2 % Ntiles);
    int kt = (int)(t2 / Ntiles);
    int k = kt * 32 + (l >> 4) * 8 + (j & 7) + (j >> 3) * 16;
    int n = nt * 16 + (l & 15);
    frag[idx] = (_Float16)w[(size_t)n * K + k];
}

// ---------------------------------------------------------------------------
// Conv1: 1->64, k=9, s=1. Cin=1 (K=81): tiny FLOPs, scalar kernel.
// ---------------------------------------------------------------------------
__global__ void conv1_relu_k(const float* __restrict__ x, const float* __restrict__ w,
                             const float* __restrict__ bias, _Float16* __restrict__ out)
{
    const int HO = 44, WO = 44, HI = 52, WI = 52;
    size_t idx = (size_t)blockIdx.x * blockDim.x + threadIdx.x;
    size_t total = (size_t)128 * HO * WO * 64;
    if (idx >= total) return;
    int c = (int)(idx & 63);
    size_t pix = idx >> 6;
    int b  = (int)(pix / (HO * WO));
    int r  = (int)(pix % (HO * WO));
    int oh = r / WO, ow = r % WO;
    const float* xb = x + (size_t)b * HI * WI;
    const float* wc = w + c * 81;
    float acc = bias[c];
#pragma unroll
    for (int kh = 0; kh < 9; ++kh)
#pragma unroll
        for (int kw = 0; kw < 9; ++kw)
            acc += xb[(oh + kh) * WI + (ow + kw)] * wc[kh * 9 + kw];
    out[idx] = (_Float16)(acc > 0.f ? acc : 0.f);
}

// ---------------------------------------------------------------------------
// TDM descriptor issue: copy a 2D tile of B fragments (STAGE_CHUNKS lines of
// 1KB, line stride = Ntiles KB) from global into LDS at byte offset lds_off.
// D# built per CDNA5 ISA 8.3/8.4 (count=1, type=2, data_size=8B).
// 6-arg builtin form: (g0, g1, g2, g3, g_extra, cpol).
// ---------------------------------------------------------------------------
__device__ __forceinline__ void tdm_stage_weights(const _Float16* gsrc, unsigned lds_off,
                                                  int d0stride_units /* Ntiles*128 */)
{
#if HAVE_TDM
    unsigned long long ga = (unsigned long long)(uintptr_t)gsrc;
    v4u g0;
    g0[0] = 1u;                                              // count=1, gather off
    g0[1] = lds_off;                                         // lds_addr (bytes)
    g0[2] = (unsigned)(ga & 0xFFFFFFFFu);                    // global_addr[31:0]
    g0[3] = (unsigned)((ga >> 32) & 0x01FFFFFFu) | 0x80000000u; // ga[56:32] | type=2
    v8i g1;
    g1[0] = 3 << 16;                   // wg_mask=0, data_size=3 (8B), no pad/iter
    g1[1] = (int)(128u << 16);         // tensor_dim0[15:0]=128 in [31:16]
    g1[2] = (int)((unsigned)STAGE_CHUNKS << 16); // tensor_dim0 hi=0 | tensor_dim1 lo=9
    g1[3] = (int)(128u << 16);         // tensor_dim1 hi=0 | tile_dim0=128
    g1[4] = STAGE_CHUNKS;              // tile_dim1=9 | tile_dim2=0
    g1[5] = d0stride_units;            // tensor_dim0_stride[31:0] (8B units)
    g1[6] = 0;                         // d0stride hi | d1stride lo
    g1[7] = 0;
    v4i z4 = {0, 0, 0, 0};
    v8i z8 = {0, 0, 0, 0, 0, 0, 0, 0};
    __builtin_amdgcn_tensor_load_to_lds(g0, g1, z4, z4, z8, 0);
#else
    (void)gsrc; (void)lds_off; (void)d0stride_units;
#endif
}

__device__ __forceinline__ void tdm_wait0()
{
#if HAVE_TDM
    __builtin_amdgcn_s_wait_tensorcnt((short)0);
#endif
}

// ---------------------------------------------------------------------------
// Implicit-GEMM conv via V_WMMA_F32_16X16X32_F16.
// Block = 8 waves = 8 M-tiles x 1 N-tile; B fragments staged to LDS by TDM
// (double buffered), shared by all 8 waves. Requires Mtiles%8==0, KT%9==0.
// ---------------------------------------------------------------------------
__global__ void conv_wmma_tdm_k(const _Float16* __restrict__ in,
                                const _Float16* __restrict__ wfrag,
                                const float* __restrict__ bias, _Float16* __restrict__ out,
                                int Hin, int Win, int Cin,
                                int Hout, int Wout, int Cout, int stride, int relu)
{
    extern __shared__ _Float16 sbuf[];   // 2 * STAGE_BYTES
    const int lane  = threadIdx.x & 31;
    const int wave  = threadIdx.x >> 5;
    const int HWo   = Hout * Wout;
    const int Ntiles = Cout >> 4;
    const int mtG = blockIdx.x / Ntiles;
    const int nt  = blockIdx.x % Ntiles;
    const int mt  = mtG * 8 + wave;
    const int khalf = lane >> 4;
    const int klo   = khalf << 3;

    // A-row pixel for this lane (A layout: lane holds row m = lane&15)
    int m  = (mt << 4) + (lane & 15);
    int b  = m / HWo, r = m % HWo;
    int oh = r / Wout, ow = r % Wout;
    const _Float16* arow = in + (((size_t)b * Hin + oh * stride) * Win + ow * stride) * Cin;

    const int CT = Cin >> 5;
    const int KT = 81 * CT;
    const int nsteps = KT / STAGE_CHUNKS;
    const int d0stride = Ntiles * 128;   // 8-byte units between k-chunk lines

    v8f acc = {};
    int buf = 0;
    int kh = 0, kw = 0, ct = 0;

    if (wave == 0) {
        tdm_stage_weights(wfrag + (size_t)nt * 512, 0, d0stride);
        tdm_wait0();
    }
    __syncthreads();

    for (int st = 0; st < nsteps; ++st) {
        if (wave == 0 && st + 1 < nsteps)
            tdm_stage_weights(wfrag + ((size_t)(st + 1) * STAGE_CHUNKS * Ntiles + nt) * 512,
                              (unsigned)((buf ^ 1) * STAGE_BYTES), d0stride);
        const _Float16* sb = sbuf + (size_t)buf * (STAGE_BYTES / 2);
#if !HAVE_TDM
        const _Float16* gb = wfrag + ((size_t)st * STAGE_CHUNKS * Ntiles + nt) * 512;
#endif
#pragma unroll
        for (int c = 0; c < STAGE_CHUNKS; ++c) {
            const _Float16* p = arow + ((size_t)kh * Win + kw) * Cin + ct * 32;
            v8h alo = *(const v8h*)(p + klo);
            v8h ahi = *(const v8h*)(p + klo + 16);
            v16h a;
#pragma unroll
            for (int j = 0; j < 8; ++j) { a[j] = alo[j]; a[j + 8] = ahi[j]; }
#if HAVE_TDM
            v16h bf = *(const v16h*)(sb + ((size_t)c * 32 + lane) * 16);
#else
            v16h bf = *(const v16h*)(gb + ((size_t)c * Ntiles * 32 + lane) * 16);
#endif
            acc = __builtin_amdgcn_wmma_f32_16x16x32_f16(
                    false, a, false, bf, (short)0, acc, false, false);
            if (++ct == CT) { ct = 0; if (++kw == 9) { kw = 0; ++kh; } }
        }
        if (wave == 0 && st + 1 < nsteps) tdm_wait0();
        __syncthreads();
        buf ^= 1;
    }

    // C/D layout: VGPR rr -> M = rr + 8*khalf, N = lane&15.
    // NHWC store address is simply pixel*Cout + ncol.
    const int ncol = (nt << 4) + (lane & 15);
    const float bv = bias[ncol];
#pragma unroll
    for (int rr = 0; rr < 8; ++rr) {
        int mm = (mt << 4) + rr + (khalf << 3);
        float vv = acc[rr] + bv;
        if (relu && vv < 0.f) vv = 0.f;
        out[(size_t)mm * Cout + ncol] = (_Float16)vv;
    }
}

// ---------------------------------------------------------------------------
// Dense GEMM (decoder) via WMMA: C[M,N] = act(A[M,K] * W[K,N] + bias)
// act: 0=none 1=relu 2=sigmoid
// ---------------------------------------------------------------------------
__global__ void gemm_wmma_k(const _Float16* __restrict__ A, const _Float16* __restrict__ wfrag,
                            const float* __restrict__ bias, _Float16* __restrict__ outh,
                            float* __restrict__ outf, int M, int K, int N, int act)
{
    const int lane = threadIdx.x & 31;
    const int wave = threadIdx.x >> 5;
    const int Mtiles = M >> 4, Ntiles = N >> 4;
    const int tile = blockIdx.x * (blockDim.x >> 5) + wave;
    if (tile >= Mtiles * Ntiles) return;
    const int mt = tile / Ntiles, nt = tile % Ntiles;
    const int khalf = lane >> 4, klo = khalf << 3;
    const _Float16* arow = A + (size_t)((mt << 4) + (lane & 15)) * K;
    const int KT = K >> 5;
    v8f acc = {};
    for (int kt = 0; kt < KT; ++kt) {
        v8h alo = *(const v8h*)(arow + kt * 32 + klo);
        v8h ahi = *(const v8h*)(arow + kt * 32 + klo + 16);
        v16h a;
#pragma unroll
        for (int j = 0; j < 8; ++j) { a[j] = alo[j]; a[j + 8] = ahi[j]; }
        v16h bf = *(const v16h*)(wfrag + ((size_t)(kt * Ntiles + nt) * 32 + lane) * 16);
        acc = __builtin_amdgcn_wmma_f32_16x16x32_f16(
                false, a, false, bf, (short)0, acc, false, false);
    }
    const int ncol = (nt << 4) + (lane & 15);
    const float bv = bias[ncol];
#pragma unroll
    for (int rr = 0; rr < 8; ++rr) {
        int mm = (mt << 4) + rr + (khalf << 3);
        float vv = acc[rr] + bv;
        if (act == 1)      vv = vv > 0.f ? vv : 0.f;
        else if (act == 2) vv = 1.f / (1.f + expf(-vv));
        if (outh) outh[(size_t)mm * N + ncol] = (_Float16)vv;
        if (outf) outf[(size_t)mm * N + ncol] = vv;
    }
}

// ---------------------------------------------------------------------------
// Primary caps reorder + squash (faithful to the torch stack/view order).
// ---------------------------------------------------------------------------
__global__ void squash_u_k(const _Float16* __restrict__ pout, float* __restrict__ u)
{
    int tid = blockIdx.x * blockDim.x + threadIdx.x;   // b*3200 + i
    if (tid >= 128 * 3200) return;
    int b = tid / 3200, i = tid % 3200;
    float s[8]; float sq = 0.f;
#pragma unroll
    for (int n = 0; n < 8; ++n) {
        int flat = i * 8 + n;
        int w = flat % 10; int t = flat / 10;
        int h = t % 10;    t /= 10;
        int g = t % 32;    int d = t / 32;
        float val = (float)pout[(((size_t)b * 10 + h) * 10 + w) * 256 + g * 8 + d];
        s[n] = val; sq += val * val;
    }
    float sc = sq / ((1.f + sq) * sqrtf(sq));
#pragma unroll
    for (int n = 0; n < 8; ++n) u[(size_t)tid * 8 + n] = s[n] * sc;
}

// u_hat[b,i,j,m] = sum_n W[i,j,n,m] * u[b,i,n]  -> stored f16 (halves the
// dominant routing traffic: 105MB instead of 210MB per pass).
__global__ void uhat_k(const float* __restrict__ u, const float* __restrict__ W,
                       _Float16* __restrict__ uhat)
{
    __shared__ float Ws[128];                 // W[i,j,:,:] (8x16)
    int ij = blockIdx.x; int i = ij >> 3; int j = ij & 7;
    int b = threadIdx.x;
    Ws[threadIdx.x] = W[((size_t)i * 8 + j) * 128 + threadIdx.x];
    __syncthreads();
    float uv[8];
#pragma unroll
    for (int n = 0; n < 8; ++n) uv[n] = u[((size_t)b * 3200 + i) * 8 + n];
    _Float16* up = uhat + (((size_t)b * 3200 + i) * 8 + j) * 16;
#pragma unroll
    for (int m = 0; m < 16; ++m) {
        float acc = 0.f;
#pragma unroll
        for (int n = 0; n < 8; ++n) acc += uv[n] * Ws[n * 16 + m];
        up[m] = (_Float16)acc;
    }
}

__global__ void zero_f32_k(float* p, size_t n)
{
    size_t i = (size_t)blockIdx.x * blockDim.x + threadIdx.x;
    if (i < n) p[i] = 0.f;
}

// c = softmax_j(b). Thread per (batch,i).
__global__ void softmax_j_k(const float* __restrict__ bb, float* __restrict__ cc)
{
    int tid = blockIdx.x * blockDim.x + threadIdx.x;
    if (tid >= 128 * 3200) return;
    const float* bp = bb + (size_t)tid * 8;
    float mx = bp[0];
#pragma unroll
    for (int j = 1; j < 8; ++j) mx = fmaxf(mx, bp[j]);
    float e[8]; float sum = 0.f;
#pragma unroll
    for (int j = 0; j < 8; ++j) { e[j] = expf(bp[j] - mx); sum += e[j]; }
    float inv = 1.f / sum;
    float* cp = cc + (size_t)tid * 8;
#pragma unroll
    for (int j = 0; j < 8; ++j) cp[j] = e[j] * inv;
}

// s[b,j,:] = sum_i c[b,i,j]*uh[b,i,j,:]; v = squash(s). Block per (b,j).
__global__ void route_sv_k(const float* __restrict__ cc, const _Float16* __restrict__ uhat,
                           float* __restrict__ v)
{
    __shared__ float red[256 * 16];
    int b = blockIdx.x >> 3, j = blockIdx.x & 7;
    float acc[16];
#pragma unroll
    for (int m = 0; m < 16; ++m) acc[m] = 0.f;
    for (int i = threadIdx.x; i < 3200; i += 256) {
        float cij = cc[((size_t)b * 3200 + i) * 8 + j];
        const _Float16* up = uhat + (((size_t)b * 3200 + i) * 8 + j) * 16;
        v8h u0 = *(const v8h*)up;
        v8h u1 = *(const v8h*)(up + 8);
#pragma unroll
        for (int m = 0; m < 8; ++m) {
            acc[m]     += cij * (float)u0[m];
            acc[m + 8] += cij * (float)u1[m];
        }
    }
#pragma unroll
    for (int m = 0; m < 16; ++m) red[threadIdx.x * 16 + m] = acc[m];
    __syncthreads();
    for (int off = 128; off > 0; off >>= 1) {
        if ((int)threadIdx.x < off)
#pragma unroll
            for (int m = 0; m < 16; ++m)
                red[threadIdx.x * 16 + m] += red[(threadIdx.x + off) * 16 + m];
        __syncthreads();
    }
    if (threadIdx.x == 0) {
        float sq = 0.f;
#pragma unroll
        for (int m = 0; m < 16; ++m) sq += red[m] * red[m];
        float sc = sq / ((1.f + sq) * sqrtf(sq));
#pragma unroll
        for (int m = 0; m < 16; ++m) v[(size_t)blockIdx.x * 16 + m] = red[m] * sc;
    }
}

// b[b,i,j] += dot(v[b,j,:], uh[b,i,j,:]). Thread per (b,i,j).
__global__ void route_bupd_k(const _Float16* __restrict__ uhat, const float* __restrict__ v,
                             float* __restrict__ bb)
{
    int tid = blockIdx.x * blockDim.x + threadIdx.x;   // ((b*3200+i)*8+j)
    if (tid >= 128 * 3200 * 8) return;
    int j = tid & 7;
    int b = tid / (3200 * 8);
    const _Float16* up = uhat + (size_t)tid * 16;
    const float* vp = v + ((size_t)b * 8 + j) * 16;
    v8h u0 = *(const v8h*)up;
    v8h u1 = *(const v8h*)(up + 8);
    float acc = 0.f;
#pragma unroll
    for (int m = 0; m < 8; ++m)
        acc += vp[m] * (float)u0[m] + vp[m + 8] * (float)u1[m];
    bb[tid] += acc;
}

__global__ void norms_k(const float* __restrict__ v, float* __restrict__ nrm)
{
    int tid = blockIdx.x * blockDim.x + threadIdx.x;   // b*8+j
    if (tid >= 128 * 8) return;
    const float* vp = v + (size_t)tid * 16;
    float sq = 0.f;
#pragma unroll
    for (int m = 0; m < 16; ++m) sq += vp[m] * vp[m];
    nrm[tid] = sqrtf(sq);
}

// softmax over batch dim (axis 0), per class j. Block per j, thread per b.
__global__ void classes_softmax_k(const float* __restrict__ nrm, float* __restrict__ cls)
{
    __shared__ float sh[128];
    int j = blockIdx.x, b = threadIdx.x;
    float val = nrm[b * 8 + j];
    sh[b] = val; __syncthreads();
    for (int off = 64; off > 0; off >>= 1) {
        if (b < off) sh[b] = fmaxf(sh[b], sh[b + off]);
        __syncthreads();
    }
    float mx = sh[0]; __syncthreads();
    float e = expf(val - mx);
    sh[b] = e; __syncthreads();
    for (int off = 64; off > 0; off >>= 1) {
        if (b < off) sh[b] += sh[b + off];
        __syncthreads();
    }
    cls[b * 8 + j] = e / sh[0];
}

// argmax over j, one-hot mask, masked decoder input t (f16), copy v out.
__global__ void mask_k(const float* __restrict__ v, const float* __restrict__ cls,
                       float* __restrict__ out_v, float* __restrict__ out_masked,
                       _Float16* __restrict__ t)
{
    int b = threadIdx.x;          // 128 threads, 1 block
    int best = 0; float bv = cls[b * 8];
#pragma unroll
    for (int j = 1; j < 8; ++j) {
        float c = cls[b * 8 + j];
        if (c > bv) { bv = c; best = j; }
    }
#pragma unroll
    for (int j = 0; j < 8; ++j) {
        out_masked[b * 8 + j] = (j == best) ? 1.f : 0.f;
#pragma unroll
        for (int m = 0; m < 16; ++m) {
            float vv = v[((size_t)b * 8 + j) * 16 + m];
            out_v[((size_t)b * 8 + j) * 16 + m] = vv;
            t[(size_t)b * 128 + j * 16 + m] = (j == best) ? (_Float16)vv : (_Float16)0.f;
        }
    }
}

// ---------------------------------------------------------------------------
extern "C" void kernel_launch(void* const* d_in, const int* in_sizes, int n_in,
                              void* d_out, int out_size, void* d_ws, size_t ws_size,
                              hipStream_t stream)
{
    (void)in_sizes; (void)n_in; (void)out_size; (void)ws_size;
    const float* x   = (const float*)d_in[0];
    const float* cw1 = (const float*)d_in[1];
    const float* cb1 = (const float*)d_in[2];
    const float* cw2 = (const float*)d_in[3];
    const float* cb2 = (const float*)d_in[4];
    const float* cw3 = (const float*)d_in[5];
    const float* cb3 = (const float*)d_in[6];
    const float* pw  = (const float*)d_in[7];
    const float* pb  = (const float*)d_in[8];
    const float* W   = (const float*)d_in[9];
    const float* dw1 = (const float*)d_in[10];
    const float* db1 = (const float*)d_in[11];
    const float* dw2 = (const float*)d_in[12];
    const float* db2 = (const float*)d_in[13];
    const float* dw3 = (const float*)d_in[14];
    const float* db3 = (const float*)d_in[15];

    float* out_v      = (float*)d_out;          // [128,8,16]
    float* out_rec    = out_v + 16384;          // [128,2704]
    float* out_masked = out_rec + 346112;       // [128,8]

    char* wsp = (char*)d_ws;
    auto alloc = [&](size_t bytes) -> void* {
        void* r = (void*)wsp; wsp += (bytes + 255) & ~(size_t)255; return r;
    };
    _Float16* h1      = (_Float16*)alloc((size_t)128 * 44 * 44 * 64 * 2);
    _Float16* w2frag  = (_Float16*)alloc((size_t)81 * 64 * 128 * 2);
    _Float16* h2      = (_Float16*)alloc((size_t)128 * 36 * 36 * 128 * 2);
    _Float16* w3frag  = (_Float16*)alloc((size_t)81 * 128 * 256 * 2);
    _Float16* h3      = (_Float16*)alloc((size_t)128 * 28 * 28 * 256 * 2);
    _Float16* pwfrag  = (_Float16*)alloc((size_t)81 * 256 * 256 * 2);
    _Float16* pout    = (_Float16*)alloc((size_t)128 * 10 * 10 * 256 * 2);
    float*    u       = (float*)alloc((size_t)128 * 3200 * 8 * 4);
    _Float16* uhat    = (_Float16*)alloc((size_t)128 * 3200 * 8 * 16 * 2);
    float*    bb      = (float*)alloc((size_t)128 * 3200 * 8 * 4);
    float*    cc      = (float*)alloc((size_t)128 * 3200 * 8 * 4);
    float*    vbuf    = (float*)alloc((size_t)128 * 8 * 16 * 4);
    float*    nrm     = (float*)alloc((size_t)128 * 8 * 4);
    float*    cls     = (float*)alloc((size_t)128 * 8 * 4);
    _Float16* t       = (_Float16*)alloc((size_t)128 * 128 * 2);
    _Float16* dw1frag = (_Float16*)alloc((size_t)128 * 512 * 2);
    _Float16* dh1     = (_Float16*)alloc((size_t)128 * 512 * 2);
    _Float16* dw2frag = (_Float16*)alloc((size_t)512 * 1024 * 2);
    _Float16* dh2     = (_Float16*)alloc((size_t)128 * 1024 * 2);
    _Float16* dw3frag = (_Float16*)alloc((size_t)1024 * 2704 * 2);

    const int TB = 256;
    auto gr = [](size_t n, int tb) { return (int)((n + tb - 1) / tb); };

    // weight repacks
    repack_conv_w_k<<<gr((size_t)81 * 64 * 128, TB),   TB, 0, stream>>>(cw2, w2frag, 64, 128);
    repack_conv_w_k<<<gr((size_t)81 * 128 * 256, TB),  TB, 0, stream>>>(cw3, w3frag, 128, 256);
    repack_conv_w_k<<<gr((size_t)81 * 256 * 256, TB),  TB, 0, stream>>>(pw,  pwfrag, 256, 256);
    repack_dense_w_k<<<gr((size_t)128 * 512, TB),      TB, 0, stream>>>(dw1, dw1frag, 128, 512);
    repack_dense_w_k<<<gr((size_t)512 * 1024, TB),     TB, 0, stream>>>(dw2, dw2frag, 512, 1024);
    repack_dense_w_k<<<gr((size_t)1024 * 2704, TB),    TB, 0, stream>>>(dw3, dw3frag, 1024, 2704);

    // conv stack (8 M-tiles per block, TDM-staged B fragments, 18KB dynamic LDS)
    conv1_relu_k<<<gr((size_t)128 * 44 * 44 * 64, TB), TB, 0, stream>>>(x, cw1, cb1, h1);
    {
        int Mtiles = 128 * 36 * 36 / 16, Ntiles = 128 / 16;
        conv_wmma_tdm_k<<<(Mtiles / 8) * Ntiles, TB, 2 * STAGE_BYTES, stream>>>(
            h1, w2frag, cb2, h2, 44, 44, 64, 36, 36, 128, 1, 1);
    }
    {
        int Mtiles = 128 * 28 * 28 / 16, Ntiles = 256 / 16;
        conv_wmma_tdm_k<<<(Mtiles / 8) * Ntiles, TB, 2 * STAGE_BYTES, stream>>>(
            h2, w3frag, cb3, h3, 36, 36, 128, 28, 28, 256, 1, 1);
    }
    {
        int Mtiles = 128 * 10 * 10 / 16, Ntiles = 256 / 16;
        conv_wmma_tdm_k<<<(Mtiles / 8) * Ntiles, TB, 2 * STAGE_BYTES, stream>>>(
            h3, pwfrag, pb, pout, 28, 28, 256, 10, 10, 256, 2, 0);
    }

    // capsules + routing
    squash_u_k<<<gr((size_t)128 * 3200, TB), TB, 0, stream>>>(pout, u);
    uhat_k<<<3200 * 8, 128, 0, stream>>>(u, W, uhat);
    zero_f32_k<<<gr((size_t)128 * 3200 * 8, TB), TB, 0, stream>>>(bb, (size_t)128 * 3200 * 8);
    for (int it = 0; it < 2; ++it) {
        softmax_j_k<<<gr((size_t)128 * 3200, TB), TB, 0, stream>>>(bb, cc);
        route_sv_k<<<128 * 8, 256, 0, stream>>>(cc, uhat, vbuf);
        route_bupd_k<<<gr((size_t)128 * 3200 * 8, TB), TB, 0, stream>>>(uhat, vbuf, bb);
    }
    softmax_j_k<<<gr((size_t)128 * 3200, TB), TB, 0, stream>>>(bb, cc);
    route_sv_k<<<128 * 8, 256, 0, stream>>>(cc, uhat, vbuf);

    // classification head + mask
    norms_k<<<gr((size_t)128 * 8, TB), TB, 0, stream>>>(vbuf, nrm);
    classes_softmax_k<<<8, 128, 0, stream>>>(nrm, cls);
    mask_k<<<1, 128, 0, stream>>>(vbuf, cls, out_v, out_masked, t);

    // decoder MLP (WMMA)
    gemm_wmma_k<<<((8 * 32) + 7) / 8,  TB, 0, stream>>>(t,   dw1frag, db1, dh1, nullptr,
                                                        128, 128, 512, 1);
    gemm_wmma_k<<<((8 * 64) + 7) / 8,  TB, 0, stream>>>(dh1, dw2frag, db2, dh2, nullptr,
                                                        128, 512, 1024, 1);
    gemm_wmma_k<<<((8 * 169) + 7) / 8, TB, 0, stream>>>(dh2, dw3frag, db3, nullptr, out_rec,
                                                        128, 1024, 2704, 2);
}
